// ASRModel_47639777247254
// MI455X (gfx1250) — matrix operand order repaired and checked by reading, not compiled, not run on hardware
//
// CDNA5 / gfx1250 implementation of the ASR block.
//
// Roofline (MI455X): ~83 GFLOP, GEMM-dominated (vocab proj 42 GFLOP,
// mamba in/out-proj ~26 GFLOP, QKV 6.4 GFLOP...). Weights ~30MB live in the
// 192MB L2; activation traffic is a few hundred MB vs 23.3 TB/s HBM, so
// compute must ride the matrix pipes: v_wmma_f32_16x16x32_bf16, f32 accum.
//
// GEMM hot loop (verified in disasm): double-buffered async global->LDS
// A-tile staging (global_load_async_to_lds_b128, ASYNCcnt-pipelined: tile
// k+1 in flight while tile k computes), ds_load_b128 fragments (operands
// pre-converted to bf16 once, zero-padded to K%64==0 -> no converts, no
// K-guards in the loop), one 32B global load per B fragment, 8 WMMAs per
// barrier. global_prefetch_b8 hints the next W chunk.
//
// The reference's MHA attends over L=B=2 (shape quirk) -> 2x2 softmax per
// (seq,head), elementwise kernel. The Mamba scan is sequential over S=2048
// with 2048 independent (b,d) channels: one lane each, 16 states in VGPRs.

#include <hip/hip_runtime.h>
#include <hip/hip_bf16.h>
#include <math.h>

#define B_   2
#define S_   2048
#define MEL_ 80
#define DM_  512
#define DI_  1024
#define N_   16
#define KC_  4
#define R_   32
#define H_   8
#define V_   10000
#define M_   (B_ * S_)          // 4096 rows in every activation GEMM

typedef __bf16 bf16_t;
typedef __attribute__((ext_vector_type(16))) __bf16 v16bf;
typedef __attribute__((ext_vector_type(8)))  __bf16 v8bf;
typedef __attribute__((ext_vector_type(8)))  float  v8f;
typedef __attribute__((ext_vector_type(4)))  int    v4i;
typedef __attribute__((address_space(3)))    v4i    lds_v4i;

// gfx1250 async global->LDS path (probe-confirmed: builtin exists, takes
// int4* pointers; emits global_load_async_to_lds_b128 + s_wait_asynccnt).
#if __has_builtin(__builtin_amdgcn_global_load_async_to_lds_b128)
#define ASYNC_LDS 1
#if __has_builtin(__builtin_amdgcn_s_wait_asynccnt)
#define WAIT_ASYNC(n) __builtin_amdgcn_s_wait_asynccnt(n)
#else
#define WAIT_ASYNC(n) asm volatile("s_wait_asynccnt %0" ::"i"(n) : "memory")
#endif
#endif

// ---------------------------------------------------------------------------
// f32 (rows x Kd, leading dim lda) -> bf16 (rows x Kdp), zero-padded K tail.
// ---------------------------------------------------------------------------
__global__ __launch_bounds__(256) void cvt_bf16_kernel(
    const float* __restrict__ in, int lda, bf16_t* __restrict__ out,
    int rows, int Kd, int Kdp)
{
  int idx = blockIdx.x * 256 + threadIdx.x;
  if (idx >= rows * Kdp) return;
  int row = idx / Kdp, col = idx - row * Kdp;
  float v = (col < Kd) ? in[(size_t)row * lda + col] : 0.f;
  out[(size_t)row * Kdp + col] = (bf16_t)v;
}

// ---------------------------------------------------------------------------
// WMMA GEMM (bf16 operands): C[M,Nn] = act( A[M,Kdp] * W[Nn,Kdp]^T + bias )
// 256 threads = 8 waves; block tile 64(M) x 128(N); wave w owns one 16-col
// N-tile and all four 16-row M sub-tiles (B fragment amortized 4x).
// A tile (64x64 bf16) double-buffered in LDS, row stride 72 elems (144B:
// 16B-aligned, bank-conflict-free). act: 0=none, 1=softplus.
// flipS: output row (b,t)->(b,S-1-t).
// ---------------------------------------------------------------------------
#define KSTEP   64
#define ASTRIDE 72

__global__ __launch_bounds__(256) void gemm_bf16_wmma_kernel(
    const bf16_t* __restrict__ A, int ldap,
    const bf16_t* __restrict__ W, int ldwp,
    const float* __restrict__ bias,
    float* __restrict__ C, int ldc,
    int Nn, int Kdp, int act, int flipS)
{
  __shared__ bf16_t As[2][64 * ASTRIDE];
  const int tid  = threadIdx.x;
  const int wave = tid >> 5;
  const int lane = tid & 31;
  const int r    = lane & 15;
  const int half = lane >> 4;
  const int m0   = blockIdx.y * 64;
  const int col0 = blockIdx.x * 128 + wave * 16;
  const bool active = (col0 < Nn);   // uniform within a wave -> EXEC all-ones

  // staging: each thread moves 16 bf16 (32B): row = tid/4, segment = tid%4
  const int srow = tid >> 2;
  const int sseg = (tid & 3) * 16;
  const bf16_t* Abase = A + (size_t)(m0 + srow) * ldap + sseg;

  // issue the (two b128) transfers of one 64x64 tile chunk for this thread
  auto stage = [&](int k0, int buf) {
    const bf16_t* gsrc = Abase + k0;
    bf16_t*       ldst = &As[buf][srow * ASTRIDE + sseg];
#if defined(ASYNC_LDS)
    __builtin_amdgcn_global_load_async_to_lds_b128(
        (v4i*)gsrc, (lds_v4i*)ldst, 0, 0);
    __builtin_amdgcn_global_load_async_to_lds_b128(
        (v4i*)(gsrc + 8), (lds_v4i*)(ldst + 8), 0, 0);
#else
    v8bf lo = *(const v8bf*)gsrc;
    v8bf hi = *(const v8bf*)(gsrc + 8);
    *(v8bf*)ldst       = lo;
    *(v8bf*)(ldst + 8) = hi;
#endif
  };

  v8f accs[4] = {};
  const int nk = Kdp / KSTEP;

  stage(0, 0);                                   // prologue: tile 0 -> buf 0
  for (int k = 0; k < nk; ++k) {
    const int cur = k & 1;
    const bool more = (k + 1 < nk);
    if (more) stage((k + 1) * KSTEP, cur ^ 1);   // tile k+1 -> other buffer
#if defined(ASYNC_LDS)
    // async loads complete in order: <=2 outstanding means tile k has landed
    // (tile k+1's two transfers may still be in flight).
    if (more) { WAIT_ASYNC(2); } else { WAIT_ASYNC(0); }
#endif
    // One barrier per iteration: it makes tile k visible to all waves. A wave writes buffer
    // b right after its own compute on b, with no barrier in between, so slower waves may
    // still be reading buffer b (this WAR hazard is not guarded).
    __syncthreads();

    if (active) {
      const int k0 = k * KSTEP;
      if (more)   // global_prefetch_b8 hint: next K chunk of W
        __builtin_prefetch(W + (size_t)(col0 + r) * ldwp + k0 + KSTEP, 0, 1);
#pragma unroll
      for (int kc = 0; kc < KSTEP; kc += 32) {
        // B fragment (ISA 16-bit 32x16): half h -> K = h*8+{0..7}, 16+h*8+{0..7}
        v16bf fb;
        {
          const bf16_t* wp = W + (size_t)(col0 + r) * ldwp + k0 + kc;
          v8bf lo = *(const v8bf*)(wp + half * 8);
          v8bf hi = *(const v8bf*)(wp + 16 + half * 8);
          fb = __builtin_shufflevector(lo, hi, 0, 1, 2, 3, 4, 5, 6, 7,
                                       8, 9, 10, 11, 12, 13, 14, 15);
        }
#pragma unroll
        for (int sub = 0; sub < 4; ++sub) {
          // A fragment (ISA 16-bit 16x32): half h -> K = h*8+{0..7}, 16+h*8+{0..7}
          const bf16_t* ap = &As[cur][(sub * 16 + r) * ASTRIDE + kc];
          v8bf lo = *(const v8bf*)(ap + half * 8);
          v8bf hi = *(const v8bf*)(ap + 16 + half * 8);
          v16bf fa = __builtin_shufflevector(lo, hi, 0, 1, 2, 3, 4, 5, 6, 7,
                                             8, 9, 10, 11, 12, 13, 14, 15);
          accs[sub] = __builtin_amdgcn_wmma_f32_16x16x32_bf16(
              false, fa, false, fb, (short)0, accs[sub], false, false);
        }
      }
    }
  }

  if (!active) return;
  const int   c  = lane & 15;
  const float bv = bias ? bias[col0 + c] : 0.f;
#pragma unroll
  for (int sub = 0; sub < 4; ++sub) {
#pragma unroll
    for (int v = 0; v < 8; ++v) {
      // C/D layout: VGPR v, lanes 0-15 -> M=v, lanes 16-31 -> M=v+8, N=lane&15
      int rr = m0 + sub * 16 + v + half * 8;
      float val = accs[sub][v] + bv;
      if (act == 1) val = (val > 20.f) ? val : log1pf(__expf(val)); // softplus
      int rout = rr;
      if (flipS) { int bb = rr / S_; int tt = rr - bb * S_; rout = bb * S_ + (S_ - 1 - tt); }
      C[(size_t)rout * ldc + col0 + c] = val;
    }
  }
}

// ---------------------------------------------------------------------------
// 2x2 attention per (n, h): one wave per pair; lane covers d = lane, lane+32.
// ---------------------------------------------------------------------------
__global__ __launch_bounds__(256) void attn_kernel(
    const float* __restrict__ qkv, float* __restrict__ o)
{
  const int wave = threadIdx.x >> 5;
  const int lane = threadIdx.x & 31;
  const int pair = blockIdx.x * 8 + wave;   // 0 .. S_*H_-1
  const int n = pair / H_;
  const int h = pair % H_;
  const int hd = DM_ / H_;                  // 64

  size_t base0 = (size_t)(0 * S_ + n) * (3 * DM_) + h * hd;
  size_t base1 = (size_t)(1 * S_ + n) * (3 * DM_) + h * hd;
  float q0a = qkv[base0 + lane],            q0b = qkv[base0 + lane + 32];
  float q1a = qkv[base1 + lane],            q1b = qkv[base1 + lane + 32];
  float k0a = qkv[base0 + DM_ + lane],      k0b = qkv[base0 + DM_ + lane + 32];
  float k1a = qkv[base1 + DM_ + lane],      k1b = qkv[base1 + DM_ + lane + 32];
  float v0a = qkv[base0 + 2 * DM_ + lane],  v0b = qkv[base0 + 2 * DM_ + lane + 32];
  float v1a = qkv[base1 + 2 * DM_ + lane],  v1b = qkv[base1 + 2 * DM_ + lane + 32];

  float s00 = q0a * k0a + q0b * k0b;
  float s01 = q0a * k1a + q0b * k1b;
  float s10 = q1a * k0a + q1b * k0b;
  float s11 = q1a * k1a + q1b * k1b;
#pragma unroll
  for (int m = 16; m >= 1; m >>= 1) {
    s00 += __shfl_xor(s00, m, 32);
    s01 += __shfl_xor(s01, m, 32);
    s10 += __shfl_xor(s10, m, 32);
    s11 += __shfl_xor(s11, m, 32);
  }
  const float inv = 0.125f;                 // 1/sqrt(64)
  s00 *= inv; s01 *= inv; s10 *= inv; s11 *= inv;
  float mx0 = fmaxf(s00, s01), mx1 = fmaxf(s10, s11);
  float e00 = __expf(s00 - mx0), e01 = __expf(s01 - mx0);
  float e10 = __expf(s10 - mx1), e11 = __expf(s11 - mx1);
  float a00 = e00 / (e00 + e01), a01 = e01 / (e00 + e01);
  float a10 = e10 / (e10 + e11), a11 = e11 / (e10 + e11);

  size_t ob0 = (size_t)(0 * S_ + n) * DM_ + h * hd;
  size_t ob1 = (size_t)(1 * S_ + n) * DM_ + h * hd;
  o[ob0 + lane]      = a00 * v0a + a01 * v1a;
  o[ob0 + lane + 32] = a00 * v0b + a01 * v1b;
  o[ob1 + lane]      = a10 * v0a + a11 * v1a;
  o[ob1 + lane + 32] = a10 * v0b + a11 * v1b;
}

// ---------------------------------------------------------------------------
// out = LN(x + res) * g + b   (one block per row, D = 512)
// ---------------------------------------------------------------------------
__global__ __launch_bounds__(256) void ln_residual_kernel(
    const float* __restrict__ x, const float* __restrict__ res,
    const float* __restrict__ g, const float* __restrict__ b,
    float* __restrict__ out, int D)
{
  __shared__ float red0[8], red1[8];
  const int row = blockIdx.x, tid = threadIdx.x;
  const float* xr = x   + (size_t)row * D;
  const float* rr = res + (size_t)row * D;
  float s = 0.f, sq = 0.f;
  for (int c = tid; c < D; c += 256) {
    float v = xr[c] + rr[c];
    s += v; sq += v * v;
  }
#pragma unroll
  for (int m = 16; m >= 1; m >>= 1) { s += __shfl_xor(s, m, 32); sq += __shfl_xor(sq, m, 32); }
  const int wave = tid >> 5, lane = tid & 31;
  if (lane == 0) { red0[wave] = s; red1[wave] = sq; }
  __syncthreads();
  if (wave == 0) {
    s  = (lane < 8) ? red0[lane] : 0.f;
    sq = (lane < 8) ? red1[lane] : 0.f;
#pragma unroll
    for (int m = 4; m >= 1; m >>= 1) { s += __shfl_xor(s, m, 32); sq += __shfl_xor(sq, m, 32); }
    if (lane == 0) { red0[0] = s; red1[0] = sq; }
  }
  __syncthreads();
  float mu  = red0[0] / D;
  float var = red1[0] / D - mu * mu;
  float rs  = rsqrtf(var + 1e-5f);
  for (int c = tid; c < D; c += 256) {
    float v = xr[c] + rr[c];
    out[(size_t)row * D + c] = (v - mu) * rs * g[c] + b[c];
  }
}

// ---------------------------------------------------------------------------
// Causal depthwise conv (K=4) + SiLU:  xc[b,t,d] from xa = xz[:, 0:DI]
// ---------------------------------------------------------------------------
__global__ __launch_bounds__(256) void conv_silu_kernel(
    const float* __restrict__ xz, const float* __restrict__ cw,
    const float* __restrict__ cb, float* __restrict__ xc)
{
  int idx = blockIdx.x * 256 + threadIdx.x;
  if (idx >= M_ * DI_) return;
  int m = idx / DI_, d = idx - m * DI_;
  int b = m / S_,    t = m - b * S_;
  float sum = cb[d];
#pragma unroll
  for (int j = 0; j < KC_; ++j) {
    int tt = t + j - (KC_ - 1);
    if (tt >= 0)
      sum += xz[(size_t)(b * S_ + tt) * (2 * DI_) + d] * cw[d * KC_ + j];
  }
  xc[(size_t)m * DI_ + d] = sum / (1.f + __expf(-sum));
}

// out[b, S-1-t, :] = in[b, t, :]
__global__ __launch_bounds__(256) void flip_kernel(
    const float* __restrict__ in, float* __restrict__ out, int D)
{
  int idx = blockIdx.x * 256 + threadIdx.x;
  if (idx >= M_ * D) return;
  int m = idx / D, c = idx - m * D;
  int b = m / S_,  t = m - b * S_;
  out[(size_t)(b * S_ + (S_ - 1 - t)) * D + c] = in[idx];
}

// ---------------------------------------------------------------------------
// Selective scan: one lane per (b,d), 16 states in VGPRs, v_exp per step.
// Writes ym = (scan_y + xc*D) * silu(z).
// ---------------------------------------------------------------------------
__global__ __launch_bounds__(256) void mamba_scan_kernel(
    const float* __restrict__ dtl,   // (M_, DI_) softplus'd dt
    const float* __restrict__ xc,    // (M_, DI_)
    const float* __restrict__ dbl,   // (M_, 64): [R_,R_+N_)=B, [R_+N_,64)=C
    const float* __restrict__ xz,    // (M_, 2*DI_): z at cols DI_..
    const float* __restrict__ A_log, // (DI_, N_)
    const float* __restrict__ Dp,    // (DI_)
    float* __restrict__ ym)          // (M_, DI_)
{
  int idx = blockIdx.x * 256 + threadIdx.x;
  if (idx >= B_ * DI_) return;
  int b = idx / DI_, d = idx - b * DI_;
  float A[N_], h[N_];
#pragma unroll
  for (int n = 0; n < N_; ++n) { A[n] = -__expf(A_log[d * N_ + n]); h[n] = 0.f; }
  const float Dd = Dp[d];
  for (int t = 0; t < S_; ++t) {
    size_t m = (size_t)b * S_ + t;
    float dt = dtl[m * DI_ + d];
    float xv = xc [m * DI_ + d];
    const float* row = dbl + m * 64;
    float dx = dt * xv, y = 0.f;
#pragma unroll
    for (int n = 0; n < N_; ++n) {
      float e = __expf(dt * A[n]);
      h[n] = e * h[n] + dx * row[R_ + n];
      y   += h[n] * row[R_ + N_ + n];
    }
    float z  = xz[m * (2 * DI_) + DI_ + d];
    float sz = z / (1.f + __expf(-z));
    ym[m * DI_ + d] = (y + xv * Dd) * sz;
  }
}

// ---------------------------------------------------------------------------
// Host orchestration
// ---------------------------------------------------------------------------
extern "C" void kernel_launch(void* const* d_in, const int* in_sizes, int n_in,
                              void* d_out, int out_size, void* d_ws, size_t ws_size,
                              hipStream_t stream)
{
  (void)in_sizes; (void)n_in; (void)out_size; (void)ws_size;
  const float* mel        = (const float*)d_in[0];
  const float* mel_w      = (const float*)d_in[1];
  const float* mel_b      = (const float*)d_in[2];
  const float* attn_in_w  = (const float*)d_in[3];
  const float* attn_in_b  = (const float*)d_in[4];
  const float* attn_out_w = (const float*)d_in[5];
  const float* attn_out_b = (const float*)d_in[6];
  const float* ln1_g = (const float*)d_in[7];
  const float* ln1_b = (const float*)d_in[8];
  const float* ln2_g = (const float*)d_in[9];
  const float* ln2_b = (const float*)d_in[10];
  // fm_* at 11..19, bm_* at 20..28
  const float* bim_w = (const float*)d_in[29];
  const float* bim_b = (const float*)d_in[30];
  const float* out_w = (const float*)d_in[31];
  const float* out_b = (const float*)d_in[32];

  // ---- scratch layout (d_ws, ~107 MB) ----
  char* w = (char*)d_ws;
  float*  x0   = (float*)(w + (size_t)0);             // 8  MB  mel proj / residual
  float*  x1   = (float*)(w + ((size_t)8  << 20));    // 8  MB  post-LN1
  float*  qkv  = (float*)(w + ((size_t)16 << 20));    // 24 MB  qkv  (reused as x2)
  float*  obuf = (float*)(w + ((size_t)40 << 20));    // 16 MB  attn o / xc
  float*  opb  = (float*)(w + ((size_t)56 << 20));    // 16 MB  attn proj / dtl
  float*  dbl  = (float*)(w + ((size_t)72 << 20));    // 1  MB  low-rank dbl
  float*  xrev = (float*)(w + ((size_t)73 << 20));    // 8  MB  flipped x1
  float*  mbuf = (float*)(w + ((size_t)81 << 20));    // 8  MB  bi-merge
  bf16_t* abf  = (bf16_t*)(w + ((size_t)89 << 20));   // 8  MB  A in bf16 (max 4096x1024)
  bf16_t* wbf  = (bf16_t*)(w + ((size_t)97 << 20));   // 10 MB  W in bf16 (max 10000x512)
  float*  x2   = qkv;                                 // reuse (qkv dead by then)

  // d_out is 4096*10000 f32 (~156 MB) and only written at the very end:
  // stage the three biggest transients in its head; all are fully consumed
  // before the final vocab GEMM overwrites d_out.
  char* od = (char*)d_out;
  float* xzbuf  = (float*)(od + (size_t)0);           // 32 MB (M_, 2*DI_)
  float* ymbuf  = (float*)(od + ((size_t)32 << 20));  // 16 MB (M_, DI_)
  float* concat = (float*)(od + ((size_t)48 << 20));  // 16 MB (M_, 2*DM_)
  float* logits = (float*)d_out;

  auto cvt = [&](const float* in, int lda, bf16_t* out, int rows, int Kd) {
    int Kdp = (Kd + 63) & ~63;
    int n = rows * Kdp;
    cvt_bf16_kernel<<<(n + 255) / 256, 256, 0, stream>>>(in, lda, out, rows, Kd, Kdp);
    return Kdp;
  };
  // C = act(A(f32,lda,Kd) @ Wf(f32,Nn x Kd)^T + bias); converts both to bf16.
  auto gemm = [&](const float* Af, int lda, int Kd, const float* Wf,
                  const float* bias, float* C, int ldc, int Nn, int act, int flip) {
    int Kdp = cvt(Af, lda, abf, M_, Kd);
    cvt(Wf, Kd, wbf, Nn, Kd);
    dim3 grid((Nn + 127) / 128, M_ / 64);
    gemm_bf16_wmma_kernel<<<grid, 256, 0, stream>>>(abf, Kdp, wbf, Kdp, bias,
                                                    C, ldc, Nn, Kdp, act, flip);
  };

  // 1) x0 = mel @ mel_w^T + mel_b
  gemm(mel, MEL_, MEL_, mel_w, mel_b, x0, DM_, DM_, 0, 0);
  // 2) qkv = x0 @ attn_in_w^T + attn_in_b
  gemm(x0, DM_, DM_, attn_in_w, attn_in_b, qkv, 3 * DM_, 3 * DM_, 0, 0);
  // 3) tiny 2x2 attention
  attn_kernel<<<S_ * H_ / 8, 256, 0, stream>>>(qkv, obuf);
  // 4) attn out-proj
  gemm(obuf, DM_, DM_, attn_out_w, attn_out_b, opb, DM_, DM_, 0, 0);
  // 5) x1 = LN(x0 + attn)
  ln_residual_kernel<<<M_, 256, 0, stream>>>(x0, opb, ln1_g, ln1_b, x1, DM_);

  // 6) Mamba, both directions (dir 0 = fwd, dir 1 = bwd on flipped input)
  for (int dir = 0; dir < 2; ++dir) {
    const float* const* p = (const float* const*)(d_in + (dir == 0 ? 11 : 20));
    const float* in_w   = p[0];
    const float* conv_w = p[1];
    const float* conv_b = p[2];
    const float* x_w    = p[3];
    const float* dt_w   = p[4];
    const float* dt_b   = p[5];
    const float* A_log  = p[6];
    const float* Dvec   = p[7];
    const float* outw   = p[8];
    const float* xin = x1;
    if (dir == 1) {
      flip_kernel<<<(M_ * DM_ + 255) / 256, 256, 0, stream>>>(x1, xrev, DM_);
      xin = xrev;
    }
    float* xc  = obuf;   // (M_, DI_)
    float* dtl = opb;    // (M_, DI_)
    // in-proj: xz = xin @ in_w^T                 (M_, 2*DI_)
    gemm(xin, DM_, DM_, in_w, nullptr, xzbuf, 2 * DI_, 2 * DI_, 0, 0);
    // depthwise conv + silu
    conv_silu_kernel<<<(M_ * DI_ + 255) / 256, 256, 0, stream>>>(xzbuf, conv_w, conv_b, xc);
    // dbl = xc @ x_w^T                           (M_, 64)
    gemm(xc, DI_, DI_, x_w, nullptr, dbl, R_ + 2 * N_, R_ + 2 * N_, 0, 0);
    // dt = softplus(dbl[:, :R] @ dt_w^T + dt_b)  (M_, DI_)
    gemm(dbl, R_ + 2 * N_, R_, dt_w, dt_b, dtl, DI_, DI_, 1, 0);
    // selective scan -> ym
    mamba_scan_kernel<<<(B_ * DI_ + 255) / 256, 256, 0, stream>>>(
        dtl, xc, dbl, xzbuf, A_log, Dvec, ymbuf);
    // out-proj into concat columns [dir*DM_, +DM_); bwd rows flipped back
    gemm(ymbuf, DI_, DI_, outw, nullptr, concat + dir * DM_, 2 * DM_, DM_, 0, dir);
  }

  // 7) bi-merge: m = concat @ bim_w^T + bim_b
  gemm(concat, 2 * DM_, 2 * DM_, bim_w, bim_b, mbuf, DM_, DM_, 0, 0);
  // 8) x2 = LN(x1 + m)
  ln_residual_kernel<<<M_, 256, 0, stream>>>(x1, mbuf, ln2_g, ln2_b, x2, DM_);
  // 9) logits = x2 @ out_w^T + out_b  (overwrites all of d_out, staging done)
  gemm(x2, DM_, DM_, out_w, out_b, logits, V_, V_, 0, 0);
}
